// MultiHeadAttention_66451734004322
// MI455X (gfx1250) — matrix operand order repaired and checked by reading, not compiled, not run on hardware
//
#include <hip/hip_runtime.h>

// ---------------------------------------------------------------------------
// MHA forward for gfx1250 (MI455X): f32 inputs -> bf16 WMMA compute (f32 acc).
// B=2, S=2048, D=1024, H=16, HD=64.
// Data movement: async global->LDS copies (ASYNCcnt) with double buffering in
// the GEMMs, TDM tensor_load_to_lds (TENSORcnt) for attention K tiles.
// ---------------------------------------------------------------------------

#define BB 2
#define SS 2048
#define DD 1024
#define HH 16
#define HDIM 64
#define MROWS (BB * SS)          // 4096

#if __has_builtin(__builtin_amdgcn_global_load_async_to_lds_b128) && \
    __has_builtin(__builtin_amdgcn_s_wait_asynccnt)
#define HAVE_ASYNC 1
#else
#define HAVE_ASYNC 0
#endif

#if __has_builtin(__builtin_amdgcn_tensor_load_to_lds) && \
    __has_builtin(__builtin_amdgcn_s_wait_tensorcnt)
#define HAVE_TDM 1
#else
#define HAVE_TDM 0
#endif

typedef __attribute__((ext_vector_type(16))) __bf16        v16bf;
typedef __attribute__((ext_vector_type(8)))  float         v8f;
typedef __attribute__((ext_vector_type(4)))  unsigned int  v4u;
typedef __attribute__((ext_vector_type(8)))  int           v8i;
typedef __attribute__((ext_vector_type(4)))  int           v4i;

// GCC-style vector to match the async builtin's pointer pointee exactly.
typedef __attribute__((__vector_size__(4 * sizeof(int)))) int gvec4i;

__device__ __forceinline__ v16bf make_frag(uint4 a, uint4 b) {
  union { uint4 u[2]; v16bf v; } x;
  x.u[0] = a; x.u[1] = b;
  return x.v;
}

#if HAVE_ASYNC
__device__ __forceinline__ void async_copy_b128(const __bf16* g, __bf16* l) {
  gvec4i* gg = (gvec4i*)(void*)(g);
  gvec4i* ll = (gvec4i*)(void*)(l);
  __builtin_amdgcn_global_load_async_to_lds_b128(
      (__attribute__((address_space(1))) gvec4i*)gg,
      (__attribute__((address_space(3))) gvec4i*)ll, 0, 0);
}
#endif

#if HAVE_TDM
// 2D TDM load: tile_w x tile_h bf16 elements, row stride = stride_elems.
// D# built per CDNA5 ISA (group0/group1; groups 2/3 zero).
__device__ __forceinline__ void tdm_load_tile_2d(const __bf16* g, __bf16* lds,
                                                 unsigned tile_w, unsigned tile_h,
                                                 unsigned stride_elems) {
  unsigned lds_off =
      (unsigned)(size_t)(__attribute__((address_space(3))) __bf16*)lds;
  unsigned long long ga = (unsigned long long)(size_t)g;
  v4u g0 = { 1u,                                   // count=1, no gather
             lds_off,                               // lds_addr
             (unsigned)ga,                          // global_addr[31:0]
             (unsigned)((ga >> 32) & 0x1FFFFFFull) | (2u << 30) };  // type=2
  v8i g1 = { (int)(1u << 16),                                  // data_size=2B
             (int)((tile_w & 0xFFFFu) << 16),                  // tensor_dim0 lo
             (int)(((tile_w >> 16) & 0xFFFFu) |
                   ((tile_h & 0xFFFFu) << 16)),                // td0 hi | td1 lo
             (int)(((tile_h >> 16) & 0xFFFFu) |
                   ((tile_w & 0xFFFFu) << 16)),                // td1 hi | tile_dim0
             (int)(tile_h & 0xFFFFu),                          // tile_dim1 (dim2=0)
             (int)stride_elems,                                // stride0[31:0]
             0, 0 };                                           // stride0 hi, stride1
  v4i z4 = { 0, 0, 0, 0 };
  v8i z8 = { 0, 0, 0, 0, 0, 0, 0, 0 };
  __builtin_amdgcn_tensor_load_to_lds(g0, g1, z4, z4, z8, 0);
}
#endif

// ---------------------------------------------------------------------------
// f32 -> bf16 conversion (grid-stride)
// ---------------------------------------------------------------------------
__global__ void cvt_f32_bf16(const float* __restrict__ s, __bf16* __restrict__ d, int n) {
  int i = blockIdx.x * blockDim.x + threadIdx.x;
  int stride = gridDim.x * blockDim.x;
  for (; i < n; i += stride) d[i] = (__bf16)s[i];
}

// ---------------------------------------------------------------------------
// Tiled transpose + convert: d[cols][rows] = (bf16) s[rows][cols]
// block (32,8), grid (cols/32, rows/32)
// ---------------------------------------------------------------------------
__global__ void cvt_transpose_f32_bf16(const float* __restrict__ s,
                                       __bf16* __restrict__ d,
                                       int rows, int cols) {
  __shared__ float tile[32][33];
  const int c0 = blockIdx.x * 32, r0 = blockIdx.y * 32;
  const int tx = threadIdx.x, ty = threadIdx.y;
#pragma unroll
  for (int i = 0; i < 32; i += 8)
    tile[ty + i][tx] = s[(size_t)(r0 + ty + i) * cols + c0 + tx];
  __syncthreads();
#pragma unroll
  for (int i = 0; i < 32; i += 8)
    d[(size_t)(c0 + ty + i) * rows + r0 + tx] = (__bf16)tile[tx][ty + i];
}

// ---------------------------------------------------------------------------
// GEMM: C[M,N] = A[M,K](bf16) * Wt[N,K](bf16, pre-transposed) + bias[N]
// Block tile 128x64, BK=32, 256 threads = 8 waves, each wave owns 32x32.
// Double-buffered LDS, async global->LDS staging.
// ---------------------------------------------------------------------------
template <bool BF16OUT>
__global__ void gemm_bias_wmma(const __bf16* __restrict__ A,
                               const __bf16* __restrict__ Wt,
                               const float* __restrict__ bias,
                               void* __restrict__ Cout,
                               int M, int N, int K) {
  __shared__ __align__(16) __bf16 ldsA[2][128 * 32];   // [row][k]
  __shared__ __align__(16) __bf16 ldsB[2][64 * 32];    // [n][k]

  const int tid  = threadIdx.x;
  const int w    = tid >> 5;
  const int lane = tid & 31;
  const int lr   = lane & 15;
  const int kg   = lane >> 4;
  const int wm   = w & 3;
  const int wn   = w >> 2;
  const int m0   = blockIdx.y * 128;
  const int n0   = blockIdx.x * 64;

  v8f acc[2][2] = {};

  const int ar = tid >> 2, ac = (tid & 3) * 8;   // A: rows ar and ar+64
  const int bn = tid >> 2, bc = (tid & 3) * 8;   // B: row n

  auto compute = [&](int buf) {
    v16bf afr[2], bfr[2];
#pragma unroll
    for (int mt = 0; mt < 2; ++mt) {
      const __bf16* rp = &ldsA[buf][(wm * 32 + mt * 16 + lr) * 32];
      afr[mt] = make_frag(*(const uint4*)(rp + kg * 8),
                          *(const uint4*)(rp + 16 + kg * 8));
    }
#pragma unroll
    for (int nt = 0; nt < 2; ++nt) {
      const __bf16* rp = &ldsB[buf][(wn * 32 + nt * 16 + lr) * 32 + kg * 8];
      bfr[nt] = make_frag(*(const uint4*)rp, *(const uint4*)(rp + 16));
    }
#pragma unroll
    for (int mt = 0; mt < 2; ++mt)
#pragma unroll
      for (int nt = 0; nt < 2; ++nt)
        acc[mt][nt] = __builtin_amdgcn_wmma_f32_16x16x32_bf16(
            false, afr[mt], false, bfr[nt], (short)0, acc[mt][nt], false, false);
  };

  const int T = K / 32;
#if HAVE_ASYNC
  auto issue = [&](int k0, int buf) {
    async_copy_b128(A + (size_t)(m0 + ar) * K + k0 + ac,      &ldsA[buf][ar * 32 + ac]);
    async_copy_b128(A + (size_t)(m0 + 64 + ar) * K + k0 + ac, &ldsA[buf][(64 + ar) * 32 + ac]);
    async_copy_b128(Wt + (size_t)(n0 + bn) * K + k0 + bc,     &ldsB[buf][bn * 32 + bc]);
  };
  issue(0, 0);
  for (int t = 0; t < T - 1; ++t) {
    issue((t + 1) * 32, (t & 1) ^ 1);             // prefetch next tile
    __builtin_amdgcn_s_wait_asynccnt(3);          // current tile landed (in-order)
    __syncthreads();
    compute(t & 1);
    __syncthreads();                              // reads done before reuse
  }
  __builtin_amdgcn_s_wait_asynccnt(0);
  __syncthreads();
  compute((T - 1) & 1);
#else
  for (int t = 0; t < T; ++t) {
    const int k0 = t * 32;
    uint4 a0 = *(const uint4*)(A + (size_t)(m0 + ar) * K + k0 + ac);
    uint4 a1 = *(const uint4*)(A + (size_t)(m0 + 64 + ar) * K + k0 + ac);
    uint4 b0 = *(const uint4*)(Wt + (size_t)(n0 + bn) * K + k0 + bc);
    if (t) __syncthreads();
    *(uint4*)&ldsA[0][ar * 32 + ac]        = a0;
    *(uint4*)&ldsA[0][(64 + ar) * 32 + ac] = a1;
    *(uint4*)&ldsB[0][bn * 32 + bc]        = b0;
    __syncthreads();
    compute(0);
  }
#endif

  // ---- epilogue: C layout: lane lr = col, VGPR r -> row = r + 8*kg ----
#pragma unroll
  for (int nt = 0; nt < 2; ++nt) {
    const int col = n0 + wn * 32 + nt * 16 + lr;
    const float bv = bias[col];
#pragma unroll
    for (int mt = 0; mt < 2; ++mt) {
      const int rbase = m0 + wm * 32 + mt * 16 + kg * 8;
#pragma unroll
      for (int r = 0; r < 8; ++r) {
        float v = acc[mt][nt][r] + bv;
        if (BF16OUT) ((__bf16*)Cout)[(size_t)(rbase + r) * N + col] = (__bf16)v;
        else         ((float*)Cout)[(size_t)(rbase + r) * N + col]  = v;
      }
    }
  }
}

// ---------------------------------------------------------------------------
// Flash attention: grid (S/64, H, B), 128 threads = 4 waves, 16 q-rows/wave.
// K tile staged by TDM (fallback: async LDS copy), V tile staged transposed.
// ---------------------------------------------------------------------------
__global__ void attention_wmma(const __bf16* __restrict__ Qb,
                               const __bf16* __restrict__ Kb,
                               const __bf16* __restrict__ Vb,
                               __bf16* __restrict__ Cxb) {
  __shared__ __align__(16) __bf16 ldsK[32 * 64];        // [key][hd]
  __shared__ __align__(16) __bf16 ldsVt[64 * 32];       // [hd][key] (transposed)
  __shared__ __align__(16) __bf16 ldsP[4 * 16 * 32];    // per-wave P tile

  const int tid  = threadIdx.x;
  const int w    = tid >> 5;
  const int lane = tid & 31;
  const int lr   = lane & 15;
  const int kg   = lane >> 4;
  const int qt   = blockIdx.x;
  const int h    = blockIdx.y;
  const int b    = blockIdx.z;
  const int colh = h * HDIM;

  // ---- Q fragments (2 x 16x32 bf16) straight from global ----
  const size_t rowQ = (size_t)b * SS + qt * 64 + w * 16 + lr;
  v16bf qf[2];
#pragma unroll
  for (int f = 0; f < 2; ++f) {
    const __bf16* qr = Qb + rowQ * DD + colh + f * 32;
    qf[f] = make_frag(*(const uint4*)(qr + kg * 8),
                      *(const uint4*)(qr + 16 + kg * 8));
  }

  v8f accc[4] = {};
  float mrow[8], lrow[8];
#pragma unroll
  for (int r = 0; r < 8; ++r) { mrow[r] = -3.0e38f; lrow[r] = 0.f; }

  const int srow   = tid >> 2;      // V staging: 32 key rows, 4 threads/row
  const int schunk = tid & 3;       // 16 bf16 each
  __bf16* pw = ldsP + w * 16 * 32;

  for (int kt = 0; kt < SS / 32; ++kt) {
    const size_t krow0 = (size_t)b * SS + kt * 32;
    __syncthreads();   // previous iteration finished reading ldsK/ldsVt

    // ---- stage K tile [32 keys x 64 hd] ----
#if HAVE_TDM
    if (w == 0)
      tdm_load_tile_2d(Kb + krow0 * DD + colh, ldsK, HDIM, 32, DD);
#elif HAVE_ASYNC
    {
      const int r0 = tid >> 3, c0 = (tid & 7) * 8;
      async_copy_b128(Kb + (krow0 + r0) * DD + colh + c0,      ldsK + r0 * 64 + c0);
      async_copy_b128(Kb + (krow0 + 16 + r0) * DD + colh + c0, ldsK + (16 + r0) * 64 + c0);
    }
#else
    {
      const uint4* gk = (const uint4*)(Kb + (krow0 + srow) * DD + colh + schunk * 16);
      uint4 k0v = gk[0], k1v = gk[1];
      *(uint4*)(ldsK + srow * 64 + schunk * 16)     = k0v;
      *(uint4*)(ldsK + srow * 64 + schunk * 16 + 8) = k1v;
    }
#endif

    // ---- stage V tile transposed -> [hd][key] ----
    {
      const uint4* gv = (const uint4*)(Vb + (krow0 + srow) * DD + colh + schunk * 16);
      uint4 v0v = gv[0], v1v = gv[1];
      const __bf16* vv = (const __bf16*)&v0v;
#pragma unroll
      for (int i = 0; i < 8; ++i) ldsVt[(schunk * 16 + i) * 32 + srow] = vv[i];
      vv = (const __bf16*)&v1v;
#pragma unroll
      for (int i = 0; i < 8; ++i) ldsVt[(schunk * 16 + 8 + i) * 32 + srow] = vv[i];
    }

    // ---- prefetch next K/V tiles into cache ----
    if (kt + 1 < SS / 32) {
      __builtin_prefetch(Kb + (krow0 + 32 + srow) * DD + colh + schunk * 16, 0, 1);
      __builtin_prefetch(Vb + (krow0 + 32 + srow) * DD + colh + schunk * 16, 0, 1);
    }

#if HAVE_TDM
    if (w == 0) __builtin_amdgcn_s_wait_tensorcnt(0);
#elif HAVE_ASYNC
    __builtin_amdgcn_s_wait_asynccnt(0);
#endif
    __syncthreads();

    // ---- scores: S = Q * K^T  (16 x 32) ----
    v8f sc[2] = {};
#pragma unroll
    for (int f = 0; f < 2; ++f) {
#pragma unroll
      for (int nt = 0; nt < 2; ++nt) {
        const __bf16* rp = ldsK + (nt * 16 + lr) * 64 + f * 32 + kg * 8;
        v16bf kf = make_frag(*(const uint4*)rp, *(const uint4*)(rp + 16));
        sc[nt] = __builtin_amdgcn_wmma_f32_16x16x32_bf16(
            false, qf[f], false, kf, (short)0, sc[nt], false, false);
      }
    }

    // ---- online softmax (rows live across lanes 0-15 / 16-31) ----
#pragma unroll
    for (int r = 0; r < 8; ++r) {
      float s0 = sc[0][r] * 0.125f;
      float s1 = sc[1][r] * 0.125f;
      float t = fmaxf(s0, s1);
      t = fmaxf(t, __shfl_xor(t, 1, 32));
      t = fmaxf(t, __shfl_xor(t, 2, 32));
      t = fmaxf(t, __shfl_xor(t, 4, 32));
      t = fmaxf(t, __shfl_xor(t, 8, 32));
      float mn = fmaxf(mrow[r], t);
      float alpha = __expf(mrow[r] - mn);
      mrow[r] = mn;
      float p0 = __expf(s0 - mn);
      float p1 = __expf(s1 - mn);
      float rs = p0 + p1;
      rs += __shfl_xor(rs, 1, 32);
      rs += __shfl_xor(rs, 2, 32);
      rs += __shfl_xor(rs, 4, 32);
      rs += __shfl_xor(rs, 8, 32);
      lrow[r] = lrow[r] * alpha + rs;
#pragma unroll
      for (int nt = 0; nt < 4; ++nt) accc[nt][r] *= alpha;
      const int prow = (r + kg * 8) * 32;
      pw[prow + lr]      = (__bf16)p0;
      pw[prow + 16 + lr] = (__bf16)p1;
    }
    __syncthreads();

    // ---- ctx += P * V  (4 WMMAs over hd tiles) ----
    {
      const __bf16* rp = pw + lr * 32;
      v16bf pf = make_frag(*(const uint4*)(rp + kg * 8),
                           *(const uint4*)(rp + 16 + kg * 8));
#pragma unroll
      for (int nt = 0; nt < 4; ++nt) {
        const __bf16* vp = ldsVt + (nt * 16 + lr) * 32 + kg * 8;
        v16bf vf = make_frag(*(const uint4*)vp, *(const uint4*)(vp + 16));
        accc[nt] = __builtin_amdgcn_wmma_f32_16x16x32_bf16(
            false, pf, false, vf, (short)0, accc[nt], false, false);
      }
    }
  }

  // ---- normalize and store ctx (bf16) ----
  const size_t orow0 = (size_t)b * SS + qt * 64 + w * 16 + kg * 8;
#pragma unroll
  for (int nt = 0; nt < 4; ++nt) {
    const int col = colh + nt * 16 + lr;
#pragma unroll
    for (int r = 0; r < 8; ++r) {
      Cxb[(orow0 + r) * DD + col] = (__bf16)(accc[nt][r] / lrow[r]);
    }
  }
}

// ---------------------------------------------------------------------------
// Launch
// ---------------------------------------------------------------------------
extern "C" void kernel_launch(void* const* d_in, const int* in_sizes, int n_in,
                              void* d_out, int out_size, void* d_ws, size_t ws_size,
                              hipStream_t stream) {
  const float* X  = (const float*)d_in[0];
  const float* Wq = (const float*)d_in[1];
  const float* bq = (const float*)d_in[2];
  const float* Wk = (const float*)d_in[3];
  const float* bk = (const float*)d_in[4];
  const float* Wv = (const float*)d_in[5];
  const float* bv = (const float*)d_in[6];
  const float* Wo = (const float*)d_in[7];
  const float* bo = (const float*)d_in[8];
  float* out = (float*)d_out;

  char* ws = (char*)d_ws;
  const size_t SZ_X = (size_t)MROWS * DD * 2;   // 8 MB
  const size_t SZ_W = (size_t)DD * DD * 2;      // 2 MB
  __bf16* Xb  = (__bf16*)(ws);
  __bf16* Wqt = (__bf16*)(ws + SZ_X);
  __bf16* Wkt = (__bf16*)(ws + SZ_X + SZ_W);
  __bf16* Wvt = (__bf16*)(ws + SZ_X + 2 * SZ_W);
  __bf16* Wot = (__bf16*)(ws + SZ_X + 3 * SZ_W);
  __bf16* Qb  = (__bf16*)(ws + SZ_X + 4 * SZ_W);
  __bf16* Kb  = (__bf16*)(ws + 2 * SZ_X + 4 * SZ_W);
  __bf16* Vb  = (__bf16*)(ws + 3 * SZ_X + 4 * SZ_W);
  __bf16* Cxb = (__bf16*)(ws + 4 * SZ_X + 4 * SZ_W);

  // 1) convert X to bf16; weights to transposed bf16 ([N][K])
  cvt_f32_bf16<<<2048, 256, 0, stream>>>(X, Xb, MROWS * DD);
  dim3 tgrid(DD / 32, DD / 32), tblk(32, 8);
  cvt_transpose_f32_bf16<<<tgrid, tblk, 0, stream>>>(Wq, Wqt, DD, DD);
  cvt_transpose_f32_bf16<<<tgrid, tblk, 0, stream>>>(Wk, Wkt, DD, DD);
  cvt_transpose_f32_bf16<<<tgrid, tblk, 0, stream>>>(Wv, Wvt, DD, DD);
  cvt_transpose_f32_bf16<<<tgrid, tblk, 0, stream>>>(Wo, Wot, DD, DD);

  // 2) QKV projections (bf16 out, bias fused)
  dim3 ggrid(DD / 64, MROWS / 128);
  gemm_bias_wmma<true><<<ggrid, 256, 0, stream>>>(Xb, Wqt, bq, Qb, MROWS, DD, DD);
  gemm_bias_wmma<true><<<ggrid, 256, 0, stream>>>(Xb, Wkt, bk, Kb, MROWS, DD, DD);
  gemm_bias_wmma<true><<<ggrid, 256, 0, stream>>>(Xb, Wvt, bv, Vb, MROWS, DD, DD);

  // 3) flash attention -> bf16 context
  attention_wmma<<<dim3(SS / 64, HH, BB), 128, 0, stream>>>(Qb, Kb, Vb, Cxb);

  // 4) output projection (f32 out, bias fused)
  gemm_bias_wmma<false><<<ggrid, 256, 0, stream>>>(Cxb, Wot, bo, out, MROWS, DD, DD);
}
